// SelfAttention_84138409329104
// MI455X (gfx1250) — matrix operand order repaired and checked by reading, not compiled, not run on hardware
//
#include <hip/hip_runtime.h>

// Problem constants (B, T, C, H) = (8, 1024, 768, 12), HD = 64
#define BB   8
#define TT   1024
#define CC   768
#define HH   12
#define HD   64
#define MTOT (BB*TT)     // 8192 rows
#define NQKV (3*CC)      // 2304

typedef __attribute__((ext_vector_type(16))) __bf16 v16bf;
typedef __attribute__((ext_vector_type(8)))  float  v8f;

union AF { v16bf v; unsigned short u[16]; };   // A/B fragment (16 bf16 = 8 VGPRs)
union CF { v8f   v; float          f[8];  };   // C/D fragment (8 f32 VGPRs)

// ---------------------------------------------------------------------------
// CDNA5 feature gates (device pass only; host pass parses the fallback)
// ---------------------------------------------------------------------------
#if defined(__gfx1250__) && __has_builtin(__builtin_amdgcn_tensor_load_to_lds)
#define USE_TDM 1
#else
#define USE_TDM 0
#endif

// Cross-lane xor exchange within a 16-lane DPP row: VALU (ROW_XMASK), no LDS port.
#if defined(__gfx1250__) && __has_builtin(__builtin_amdgcn_update_dpp)
#define XSWAP(x, m) __int_as_float(__builtin_amdgcn_update_dpp(0, __float_as_int(x), 0x160 | (m), 0xF, 0xF, true))
#else
#define XSWAP(x, m) __shfl_xor((x), (m))
#endif
#define RMAX16(x) do { x = fmaxf(x, XSWAP(x,1)); x = fmaxf(x, XSWAP(x,2)); \
                       x = fmaxf(x, XSWAP(x,4)); x = fmaxf(x, XSWAP(x,8)); } while (0)
#define RSUM16(x) do { x = x + XSWAP(x,1); x = x + XSWAP(x,2); \
                       x = x + XSWAP(x,4); x = x + XSWAP(x,8); } while (0)

#if USE_TDM
typedef __attribute__((ext_vector_type(4))) unsigned int u32x4;
typedef __attribute__((ext_vector_type(4))) int          i32x4;
typedef __attribute__((ext_vector_type(8))) int          i32x8;

// Flat->LDS aperture: LDS offset is the low 32 bits of the flat address (ISA 10.2 aperture table)
__device__ inline unsigned int lds_off(const void* p) {
    return (unsigned int)(unsigned long long)p;
}

// Issue one 2D-tile TENSOR_LOAD_TO_LDS (D# per ISA 8.3/8.4, 2-byte elements).
__device__ inline void tdm_load_2d(unsigned int lds_byte, const void* gptr,
                                   unsigned int tile_ex, unsigned int tile_rows,
                                   unsigned int tens_ex, unsigned int tens_rows,
                                   unsigned int stride_e)
{
    unsigned long long ga = (unsigned long long)gptr;
    u32x4 g0;
    g0[0] = 1u;                                        // count=1 (valid), user mode
    g0[1] = lds_byte;                                  // lds_addr
    g0[2] = (unsigned int)ga;                          // global_addr[31:0]
    g0[3] = (unsigned int)(ga >> 32) | (2u << 30);     // global_addr[56:32] | type=2
    i32x8 g1;
    g1[0] = (int)(1u << 16);                           // data_size=1 (2B), mask=0
    g1[1] = (int)((tens_ex & 0xFFFFu) << 16);          // tensor_dim0[15:0] @ bits 63:48
    g1[2] = (int)(((tens_ex >> 16) & 0xFFFFu) | ((tens_rows & 0xFFFFu) << 16));
    g1[3] = (int)(((tens_rows >> 16) & 0xFFFFu) | (tile_ex << 16));   // tile_dim0
    g1[4] = (int)tile_rows;                            // tile_dim1 (tile_dim2 = 0)
    g1[5] = (int)stride_e;                             // tensor_dim0_stride[31:0]
    g1[6] = 0;
    g1[7] = 0;
    i32x4 z4 = {0, 0, 0, 0};
#if __clang_major__ >= 23
    i32x8 z8 = {0, 0, 0, 0, 0, 0, 0, 0};
    __builtin_amdgcn_tensor_load_to_lds(g0, g1, z4, z4, z8, 0);
#else
    __builtin_amdgcn_tensor_load_to_lds(g0, g1, z4, z4, 0);
#endif
}
#endif // USE_TDM

__device__ inline unsigned short f2bf(float f) {
    union { float f; unsigned int u; } v; v.f = f;
    unsigned int r = v.u + 0x7FFFu + ((v.u >> 16) & 1u);   // round-to-nearest-even
    return (unsigned short)(r >> 16);
}

// A-matrix 16x32 bf16: lanes 0-15 K=0..15, lanes 16-31 K=16..31 (matches B loaders)
__device__ inline int kbase_a(int v, int hi) {
    return 2*v + (hi ? 16 : 0);
}

__device__ inline AF load_a(const unsigned short* t, int row, int stride, int kOff, int hi) {
    AF a;
#pragma unroll
    for (int v = 0; v < 8; ++v) {
        int k = kOff + kbase_a(v, hi);
        a.u[2*v]   = t[row*stride + k];
        a.u[2*v+1] = t[row*stride + k + 1];
    }
    return a;
}

// B-matrix 32x16 bf16: lanes 0-15 K=0..15, lanes 16-31 K=16..31, N = lane%16.
__device__ inline AF load_b_rm(const unsigned short* t, int n, int stride, int kOff, int hi) {
    AF a; int kb = kOff + (hi ? 16 : 0);
#pragma unroll
    for (int v = 0; v < 8; ++v) {
        a.u[2*v]   = t[(kb + 2*v)   * stride + n];
        a.u[2*v+1] = t[(kb + 2*v+1) * stride + n];
    }
    return a;
}

__device__ inline AF load_b_cm(const unsigned short* t, int n, int stride, int kOff, int hi) {
    AF a; int kb = kOff + (hi ? 16 : 0);
#pragma unroll
    for (int v = 0; v < 8; ++v) {
        a.u[2*v]   = t[n*stride + kb + 2*v];
        a.u[2*v+1] = t[n*stride + kb + 2*v + 1];
    }
    return a;
}

__global__ void cast_f32_bf16(const float* __restrict__ in, unsigned short* __restrict__ out, int n) {
    for (int i = blockIdx.x * blockDim.x + threadIdx.x; i < n; i += gridDim.x * blockDim.x)
        out[i] = f2bf(in[i]);
}

// ---------------------------------------------------------------------------
// bf16 GEMM, 64x64 WG tile, 8 waves (4M x 2N), K-step 32, TDM double-buffered.
// MODE 0: qkv = x@Wqkv + b, fused RoPE + split into Q/K/V [B,H,T,HD] bf16
// MODE 1: out = A@Wproj + b, f32 output
// ---------------------------------------------------------------------------
template<int MODE>
__global__ __launch_bounds__(256)
void gemm_bf16(const unsigned short* __restrict__ A,
               const unsigned short* __restrict__ Bw,
               const float* __restrict__ bias,
               unsigned short* __restrict__ Qb,
               unsigned short* __restrict__ Kb,
               unsigned short* __restrict__ Vb,
               float* __restrict__ Out,
               int M, int N, int K)
{
    __shared__ __align__(16) unsigned short aT[2][64*32];
    __shared__ __align__(16) unsigned short bT[2][32*64];
    const int tid = threadIdx.x, w = tid >> 5, lane = tid & 31;
    const int lr = lane & 15, hi = (lane >= 16);
    const int wm = w & 3, wn = w >> 2;
    const int m0 = blockIdx.y * 64, n0 = blockIdx.x * 64;

    CF c0, c1;
#pragma unroll
    for (int i = 0; i < 8; ++i) { c0.f[i] = 0.f; c1.f[i] = 0.f; }

    const int nsteps = K >> 5;
#if USE_TDM
    if (w == 0) {   // TDM: DMA tile 0 into buffer 0 (A: 64x32 rows of K, B: 32x64 of N)
        tdm_load_2d(lds_off(aT[0]), A  + (size_t)m0*K, 32, 64, (unsigned)K, (unsigned)M, (unsigned)K);
        tdm_load_2d(lds_off(bT[0]), Bw + (size_t)n0,   64, 32, (unsigned)N, (unsigned)K, (unsigned)N);
    }
#else
    const int arow = tid >> 2, acol = (tid & 3) * 8;   // 64x32 tile, 1 uint4/thread
    const int brow = tid >> 3, bcol = (tid & 7) * 8;   // 32x64 tile, 1 uint4/thread
#endif

    for (int i = 0; i < nsteps; ++i) {
        const int buf = i & 1;
        const int k0 = i << 5;
#if USE_TDM
        if (w == 0) {
            if (i + 1 < nsteps) {   // prefetch next tile into other buffer, overlap with WMMA
                tdm_load_2d(lds_off(aT[buf ^ 1]), A  + (size_t)m0*K + (k0 + 32), 32, 64, (unsigned)K, (unsigned)M, (unsigned)K);
                tdm_load_2d(lds_off(bT[buf ^ 1]), Bw + (size_t)(k0 + 32)*N + n0, 64, 32, (unsigned)N, (unsigned)K, (unsigned)N);
                __builtin_amdgcn_s_wait_tensorcnt(2);   // oldest 2 (tile i) complete
            } else {
                __builtin_amdgcn_s_wait_tensorcnt(0);
            }
        }
#else
        *(uint4*)&aT[buf][arow*32 + acol] = *(const uint4*)&A[(size_t)(m0 + arow)*K + k0 + acol];
        *(uint4*)&bT[buf][brow*64 + bcol] = *(const uint4*)&Bw[(size_t)(k0 + brow)*N + n0 + bcol];
        if (k0 + 32 < K) {
            __builtin_prefetch(&A[(size_t)(m0 + arow)*K + k0 + 32 + acol], 0, 0);
            __builtin_prefetch(&Bw[(size_t)(k0 + 32 + brow)*N + n0 + bcol], 0, 0);
        }
#endif
        __syncthreads();
        AF a  = load_a(aT[buf] + wm*16*32, lr, 32, 0, hi);
        AF b0 = load_b_rm(bT[buf], wn*32 + lr,      64, 0, hi);
        AF b1 = load_b_rm(bT[buf], wn*32 + 16 + lr, 64, 0, hi);
        c0.v = __builtin_amdgcn_wmma_f32_16x16x32_bf16(false, a.v, false, b0.v, (short)0, c0.v, false, false);
        c1.v = __builtin_amdgcn_wmma_f32_16x16x32_bf16(false, a.v, false, b1.v, (short)0, c1.v, false, false);
        __syncthreads();
    }

    // Epilogue. C layout: lane<16 -> (M=v, N=lane); lane>=16 -> (M=v+8, N=lane-16)
    const int rowg = m0 + wm*16 + (hi ? 8 : 0);
#pragma unroll
    for (int fIdx = 0; fIdx < 2; ++fIdx) {
        CF& c = fIdx ? c1 : c0;
        const int colg = n0 + wn*32 + fIdx*16 + lr;
        const float bval = bias[colg];
#pragma unroll
        for (int v = 0; v < 8; ++v) {
            const int row = rowg + v;
            float val = c.f[v] + bval;
            if (MODE == 0) {
                // adjacent lanes hold adjacent channels -> rotate pairs via DPP xor1
                float partner = XSWAP(val, 1);
                int sidx = colg / CC;              // 0=q, 1=k, 2=v
                int rem  = colg - sidx*CC;
                int h = rem >> 6, hd = rem & 63;
                int t = row & (TT - 1);
                int bb = row >> 10;
                float o;
                if (sidx < 2) {
                    int j = hd >> 1;               // freq = 10000^(-j/32)
                    float ang = (float)t * __expf((float)j * -0.28782313662425575f);
                    float sn, cs; __sincosf(ang, &sn, &cs);
                    o = (hd & 1) ? (partner*sn + val*cs) : (val*cs - partner*sn);
                } else {
                    o = val;
                }
                unsigned short* dst = (sidx == 0) ? Qb : ((sidx == 1) ? Kb : Vb);
                dst[(((size_t)(bb*HH + h))*TT + t)*HD + hd] = f2bf(o);
            } else {
                Out[(size_t)row*N + colg] = val;
            }
        }
    }
}

// ---------------------------------------------------------------------------
// Flash attention: grid (T/64, H, B); 128 threads = 4 waves x 16 query rows.
// ---------------------------------------------------------------------------
__global__ __launch_bounds__(128)
void attn_kernel(const unsigned short* __restrict__ Qb,
                 const unsigned short* __restrict__ Kb,
                 const unsigned short* __restrict__ Vb,
                 unsigned short* __restrict__ Ab)
{
    __shared__ __align__(16) unsigned short kT[32*HD];      // 32 keys x 64
    __shared__ __align__(16) unsigned short vT[32*HD];
    __shared__ __align__(16) unsigned short qT[64*HD];      // 64 q rows x 64
    __shared__ __align__(16) unsigned short pT[4*16*32];    // per-wave P tile

    const int tid = threadIdx.x, w = tid >> 5, lane = tid & 31;
    const int lr = lane & 15, hi = (lane >= 16);
    const int b = blockIdx.z, h = blockIdx.y, q0wg = blockIdx.x * 64;
    const size_t base = ((size_t)(b*HH + h)) * TT * HD;

#if USE_TDM
    if (w == 0) {
        tdm_load_2d(lds_off(qT), Qb + base + (size_t)q0wg*HD, HD, 64, HD, TT, HD);
        __builtin_amdgcn_s_wait_tensorcnt(0);
    }
#else
    for (int i = tid; i < 512; i += 128) {                  // 64x64 bf16 = 512 uint4
        int r = i >> 3, c = (i & 7) * 8;
        *(uint4*)&qT[r*HD + c] = *(const uint4*)&Qb[base + (size_t)(q0wg + r)*HD + c];
    }
#endif
    __syncthreads();

    const int q0 = q0wg + w*16;
    AF qa0 = load_a(qT + w*16*HD, lr, HD, 0,  hi);          // Q rows, d = 0..31
    AF qa1 = load_a(qT + w*16*HD, lr, HD, 32, hi);          // d = 32..63

    CF O[4];
    float mrow[8], lrow[8];
#pragma unroll
    for (int v = 0; v < 8; ++v) { mrow[v] = -1e30f; lrow[v] = 0.f; }
#pragma unroll
    for (int j = 0; j < 4; ++j)
#pragma unroll
        for (int v = 0; v < 8; ++v) O[j].f[v] = 0.f;

    unsigned short* pw = pT + w*16*32;
    const int nkb = (q0wg + 63) / 32 + 1;                   // causal key-block bound for WG

    for (int kb = 0; kb < nkb; ++kb) {
        const int key0 = kb * 32;
#if USE_TDM
        if (w == 0) {
            tdm_load_2d(lds_off(kT), Kb + base + (size_t)key0*HD, HD, 32, HD, TT, HD);
            tdm_load_2d(lds_off(vT), Vb + base + (size_t)key0*HD, HD, 32, HD, TT, HD);
            __builtin_amdgcn_s_wait_tensorcnt(0);
        }
#else
        for (int i = tid; i < 256; i += 128) {              // 32x64 bf16 = 256 uint4 each
            int r = i >> 2, c = (i & 3) * 8;
            *(uint4*)&kT[r*HD + c] = *(const uint4*)&Kb[base + (size_t)(key0 + r)*HD + c];
            *(uint4*)&vT[r*HD + c] = *(const uint4*)&Vb[base + (size_t)(key0 + r)*HD + c];
        }
#endif
        __syncthreads();

        const bool active = (key0 <= q0 + 15);              // wave-uniform
        CF s0, s1;
        if (active) {
#pragma unroll
            for (int v = 0; v < 8; ++v) { s0.f[v] = 0.f; s1.f[v] = 0.f; }
#pragma unroll
            for (int dc = 0; dc < 2; ++dc) {                // S = Q x K^T
                const AF& qa = dc ? qa1 : qa0;
                AF kf0 = load_b_cm(kT, 0  + lr, HD, dc*32, hi);
                AF kf1 = load_b_cm(kT, 16 + lr, HD, dc*32, hi);
                s0.v = __builtin_amdgcn_wmma_f32_16x16x32_bf16(false, qa.v, false, kf0.v, (short)0, s0.v, false, false);
                s1.v = __builtin_amdgcn_wmma_f32_16x16x32_bf16(false, qa.v, false, kf1.v, (short)0, s1.v, false, false);
            }
            // scale + causal mask (lane owns key column, vgpr owns query row)
            const int trow = q0 + (hi ? 8 : 0);
#pragma unroll
            for (int v = 0; v < 8; ++v) {
                int t = trow + v;
                float x0 = s0.f[v] * 0.125f; if (key0 + lr      > t) x0 = -1e30f;
                float x1 = s1.f[v] * 0.125f; if (key0 + 16 + lr > t) x1 = -1e30f;
                s0.f[v] = x0; s1.f[v] = x1;
            }
            // online softmax: DPP ROW_XMASK reductions within the owning 16-lane half
            float bm[8];
#pragma unroll
            for (int v = 0; v < 8; ++v) { float t = fmaxf(s0.f[v], s1.f[v]); RMAX16(t); bm[v] = t; }
#pragma unroll
            for (int v = 0; v < 8; ++v) {
                float nm = fmaxf(mrow[v], bm[v]);
                float sc = __expf(mrow[v] - nm);
                mrow[v] = nm;
                lrow[v] *= sc;
#pragma unroll
                for (int j = 0; j < 4; ++j) O[j].f[v] *= sc;
                s0.f[v] = __expf(s0.f[v] - nm);
                s1.f[v] = __expf(s1.f[v] - nm);
            }
#pragma unroll
            for (int v = 0; v < 8; ++v) { float t = s0.f[v] + s1.f[v]; RSUM16(t); lrow[v] += t; }
            // repack P (C layout) -> row-major LDS for A-fragment reload
#pragma unroll
            for (int v = 0; v < 8; ++v) {
                int row = v + (hi ? 8 : 0);
                pw[row*32 + lr]      = f2bf(s0.f[v]);
                pw[row*32 + 16 + lr] = f2bf(s1.f[v]);
            }
        }
        __syncthreads();
        if (active) {
            AF pa = load_a(pw, lr, 32, 0, hi);              // P as 16x32 A fragment
#pragma unroll
            for (int j = 0; j < 4; ++j) {
                AF vf = load_b_rm(vT, j*16 + lr, HD, 0, hi);
                O[j].v = __builtin_amdgcn_wmma_f32_16x16x32_bf16(false, pa.v, false, vf.v, (short)0, O[j].v, false, false);
            }
        }
        __syncthreads();
    }

    // normalize and store attn output bf16 [B, T, C]
#pragma unroll
    for (int j = 0; j < 4; ++j)
#pragma unroll
        for (int v = 0; v < 8; ++v) {
            int t  = q0 + v + (hi ? 8 : 0);
            int hd = j*16 + lr;
            float val = O[j].f[v] / lrow[v];
            Ab[((size_t)(b*TT + t))*CC + h*HD + hd] = f2bf(val);
        }
}

extern "C" void kernel_launch(void* const* d_in, const int* in_sizes, int n_in,
                              void* d_out, int out_size, void* d_ws, size_t ws_size,
                              hipStream_t stream)
{
    (void)in_sizes; (void)n_in; (void)out_size; (void)ws_size;
    const float* x     = (const float*)d_in[0];
    const float* Wqkv  = (const float*)d_in[1];
    const float* bqkv  = (const float*)d_in[2];
    const float* Wproj = (const float*)d_in[3];
    const float* bproj = (const float*)d_in[4];
    // d_in[5] (mask) unused: causal mask applied analytically

    char* ws = (char*)d_ws;
    size_t off = 0;
    auto carve = [&](size_t bytes) -> void* {
        void* p = ws + off;
        off = (off + bytes + 255) & ~(size_t)255;
        return p;
    };
    unsigned short* Xb     = (unsigned short*)carve((size_t)MTOT*CC*2);
    unsigned short* Wqkvb  = (unsigned short*)carve((size_t)CC*NQKV*2);
    unsigned short* Wprojb = (unsigned short*)carve((size_t)CC*CC*2);
    unsigned short* Qb     = (unsigned short*)carve((size_t)BB*HH*TT*HD*2);
    unsigned short* Kb     = (unsigned short*)carve((size_t)BB*HH*TT*HD*2);
    unsigned short* Vb     = (unsigned short*)carve((size_t)BB*HH*TT*HD*2);
    unsigned short* Ab     = (unsigned short*)carve((size_t)MTOT*CC*2);

    cast_f32_bf16<<<4096, 256, 0, stream>>>(x,     Xb,     MTOT*CC);
    cast_f32_bf16<<<4096, 256, 0, stream>>>(Wqkv,  Wqkvb,  CC*NQKV);
    cast_f32_bf16<<<2048, 256, 0, stream>>>(Wproj, Wprojb, CC*CC);

    gemm_bf16<0><<<dim3(NQKV/64, MTOT/64), 256, 0, stream>>>(
        Xb, Wqkvb, bqkv, Qb, Kb, Vb, nullptr, MTOT, NQKV, CC);

    attn_kernel<<<dim3(TT/64, HH, BB), 128, 0, stream>>>(Qb, Kb, Vb, Ab);

    gemm_bf16<1><<<dim3(CC/64, MTOT/64), 256, 0, stream>>>(
        Ab, Wprojb, bproj, nullptr, nullptr, nullptr, (float*)d_out, MTOT, CC, CC);
}
